// SelfAttention_60859686584748
// MI455X (gfx1250) — matrix operand order repaired and checked by reading, not compiled, not run on hardware
//
#include <hip/hip_runtime.h>

// Problem constants (from reference): B=8, C=256, H=W=64 -> N=4096, D=32.
#define Bb 8
#define Cc 256
#define Nn 4096
#define Dd 32
#define VSTR 40   // LDS row stride in halves (80B) -> conflict-free ds_load_b128

typedef __attribute__((ext_vector_type(16))) _Float16 v16h;
typedef __attribute__((ext_vector_type(8)))  _Float16 v8h;
typedef __attribute__((ext_vector_type(8)))  float    v8f;

#if defined(__has_builtin)
#if __has_builtin(__builtin_amdgcn_global_load_async_to_lds_b128)
#define HAVE_ASYNC_LDS 1
#endif
#endif
#ifndef HAVE_ASYNC_LDS
#define HAVE_ASYNC_LDS 0
#endif

// 16B global -> LDS copy. Async (ASYNCcnt-tracked) when available.
// Probe diagnostic showed param0 is `int __vector(4) __device__ *`, i.e. a
// pointer to a 16-byte int vector in the global address space.
__device__ __forceinline__ void copy16_to_lds(const _Float16* g, _Float16* l) {
#if HAVE_ASYNC_LDS
    typedef int v4i_ __attribute__((vector_size(16)));
    typedef __attribute__((address_space(1))) v4i_* gp_t;
    typedef __attribute__((address_space(3))) v4i_* lp_t;
    // Generic LDS pointer's low 32 bits are the LDS byte offset.
    __builtin_amdgcn_global_load_async_to_lds_b128(
        (gp_t)(uintptr_t)g, (lp_t)(uint32_t)(uintptr_t)l, 0, 0);
#else
    *(v8h*)l = *(const v8h*)g;   // global_load_b128 + ds_store_b128
#endif
}

__device__ __forceinline__ void wait_stage() {
#if HAVE_ASYNC_LDS
#if __has_builtin(__builtin_amdgcn_s_wait_asynccnt)
    __builtin_amdgcn_s_wait_asynccnt(0);
#else
    asm volatile("s_wait_asynccnt 0x0" ::: "memory");
#endif
#endif
}

// Load a 16x32 (MxK) f16 A-operand row fragment for this lane:
// K order used here (contiguous, k = 16h + s): lanes 0-15 hold K=0..7 (elems 0-7) and
// K=8..15 (elems 8-15); lanes 16-31 hold K=16..23 and K=24..31.
__device__ __forceinline__ v16h load_a_frag(const _Float16* row, int o0, int o1) {
    v8h lo = *(const v8h*)(row + o0);
    v8h hi = *(const v8h*)(row + o1);
    return __builtin_shufflevector(lo, hi, 0,1,2,3,4,5,6,7,8,9,10,11,12,13,14,15);
}

// ---------------------------------------------------------------------------
// Phase 1: 1x1-conv projections. Q,K -> [B,N,D] f16 ; V -> [B,C,N] f16.
// ~5.4 GFLOP total -> VALU is fine; stage the x column tile in LDS.
// ---------------------------------------------------------------------------
__global__ void proj_kernel(const float* __restrict__ x,
                            const float* __restrict__ wq, const float* __restrict__ bq,
                            const float* __restrict__ wk, const float* __restrict__ bk,
                            const float* __restrict__ wv, const float* __restrict__ bv,
                            _Float16* __restrict__ qws,
                            _Float16* __restrict__ kws,
                            _Float16* __restrict__ vws) {
    __shared__ float xs[Cc][16];           // 16 KB: x[b, :, n0..n0+15]
    const int blk = blockIdx.x;            // B * N/16 blocks
    const int b  = blk / (Nn / 16);
    const int n0 = (blk % (Nn / 16)) * 16;
    const int t   = threadIdx.x;           // 256 threads = 8 waves
    const int j   = t & 15;                // pixel within tile
    const int row = t >> 4;                // 0..15

    const float* xb = x + (size_t)b * Cc * Nn;
    for (int c = row; c < Cc; c += 16)
        xs[c][j] = xb[(size_t)c * Nn + (n0 + j)];
    __syncthreads();

    _Float16* vbw = vws + (size_t)b * Cc * Nn;
    for (int i = 0; i < 16; ++i) {
        const int co = row + 16 * i;
        const float* w = wv + (size_t)co * Cc;
        float a = bv[co];
        #pragma unroll 8
        for (int c = 0; c < Cc; ++c) a += w[c] * xs[c][j];
        vbw[(size_t)co * Nn + (n0 + j)] = (_Float16)a;
    }

    _Float16* qr = qws + ((size_t)b * Nn + (n0 + j)) * Dd;
    _Float16* kr = kws + ((size_t)b * Nn + (n0 + j)) * Dd;
    for (int d = row; d < Dd; d += 16) {
        const float* wqr = wq + (size_t)d * Cc;
        const float* wkr = wk + (size_t)d * Cc;
        float aq = bq[d], ak = bk[d];
        #pragma unroll 8
        for (int c = 0; c < Cc; ++c) {
            const float xv = xs[c][j];
            aq += wqr[c] * xv;
            ak += wkr[c] * xv;
        }
        qr[d] = (_Float16)aq;
        kr[d] = (_Float16)ak;
    }
}

// ---------------------------------------------------------------------------
// Phase 2: flash attention. 8 waves per workgroup, each owning a 16-query
// tile; the workgroup shares double-buffered K/V 32-key tiles in LDS, staged
// with async global->LDS copies. Per 32-key step per wave: 2 WMMA (E^T),
// online softmax, P repack via xor-16 shuffles, 16 WMMA (out^T += V^T x P).
// ---------------------------------------------------------------------------
__global__ void __launch_bounds__(256)
attn_kernel(const _Float16* __restrict__ qws,
            const _Float16* __restrict__ kws,
            const _Float16* __restrict__ vws,
            const float* __restrict__ x,
            const float* __restrict__ gamma,
            float* __restrict__ out) {
    __shared__ _Float16 ksh[2][32 * VSTR];    //  5 KB
    __shared__ _Float16 vsh[2][Cc * VSTR];    // 40 KB

    const int blk = blockIdx.x;               // B * N/128 blocks
    const int b   = blk / (Nn / 128);
    const int qb0 = (blk % (Nn / 128)) * 128;
    const int t    = threadIdx.x;             // 256 threads
    const int wave = t >> 5;                  // 0..7
    const int lane = t & 31;
    const int hf   = lane >> 4;
    const int lm   = lane & 15;
    const int q0   = qb0 + wave * 16;         // this wave's query tile

    const _Float16* qb = qws + (size_t)b * Nn * Dd;
    const _Float16* kb = kws + (size_t)b * Nn * Dd;
    const _Float16* vb = vws + (size_t)b * Cc * Nn;

    // B operand (K=32=D x N=16 queries): one contiguous 32B load per lane.
    const v16h qB = *(const v16h*)(qb + (size_t)(q0 + lm) * Dd + hf * 16);

    const int o0 = hf * 16;                   // A-frag K-offsets per half-wave
    const int o1 = hf * 16 + 8;

    v8f zero = {};
    v8f acc[16];                              // out^T: 16 c-tiles of 16x16 f32
    #pragma unroll
    for (int tt = 0; tt < 16; ++tt) acc[tt] = zero;
    float m_run = -3.0e38f;
    float l_run = 0.0f;

    // ---- tile staging: V = 256 rows x 4 chunks, K = 32 rows x 4 chunks ----
    auto stage = [&](int bi, int kt) {
        #pragma unroll
        for (int i = 0; i < 4; ++i) {
            const int lin = t + 256 * i;      // 0..1023
            const int c = lin >> 2, ch = lin & 3;
            copy16_to_lds(vb + (size_t)c * Nn + kt + ch * 8,
                          &vsh[bi][c * VSTR + ch * 8]);
        }
        if (t < 128) {
            const int r = t >> 2, ch = t & 3;
            copy16_to_lds(kb + (size_t)(kt + r) * Dd + ch * 8,
                          &ksh[bi][r * VSTR + ch * 8]);
        }
    };

    stage(0, 0);

    for (int kt = 0; kt < Nn; kt += 32) {
        const int cur = (kt >> 5) & 1;
        wait_stage();
        __syncthreads();
        if (kt + 32 < Nn) stage(cur ^ 1, kt + 32);

        const _Float16* ks = ksh[cur];
        const _Float16* vs = vsh[cur];

        // E^T tiles from LDS: rows = keys, cols = queries.
        const v16h a0 = load_a_frag(ks + lm * VSTR, o0, o1);
        const v16h a1 = load_a_frag(ks + (16 + lm) * VSTR, o0, o1);
        v8f e0 = __builtin_amdgcn_wmma_f32_16x16x32_f16(false, a0, false, qB,
                                                        (short)0, zero, false, false);
        v8f e1 = __builtin_amdgcn_wmma_f32_16x16x32_f16(false, a1, false, qB,
                                                        (short)0, zero, false, false);

        // Per-query max: 16 in-lane values + partner half via one xor-16.
        float tm = e0[0];
        #pragma unroll
        for (int r = 1; r < 8; ++r) tm = fmaxf(tm, e0[r]);
        #pragma unroll
        for (int r = 0; r < 8; ++r) tm = fmaxf(tm, e1[r]);
        tm = fmaxf(tm, __shfl_xor(tm, 16, 32));
        const float m_new = fmaxf(m_run, tm);
        const float alpha = __expf(m_run - m_new);   // 0 on first tile

        float p0[8], p1[8], s = 0.0f;
        #pragma unroll
        for (int r = 0; r < 8; ++r) { p0[r] = __expf(e0[r] - m_new); s += p0[r]; }
        #pragma unroll
        for (int r = 0; r < 8; ++r) { p1[r] = __expf(e1[r] - m_new); s += p1[r]; }
        s += __shfl_xor(s, 16, 32);
        l_run = l_run * alpha + s;
        m_run = m_new;

        #pragma unroll
        for (int tt = 0; tt < 16; ++tt) {
            #pragma unroll
            for (int r = 0; r < 8; ++r) acc[tt][r] *= alpha;
        }

        // Repack P (D-layout) into B-operand layout (32 keys x 16 queries):
        // lanes 0-15 need keys 0..15, lanes 16-31 keys 16..31; missing 8 per
        // lane live in the partner half -> xor-16 exchange.
        float x0[8], x1[8];
        #pragma unroll
        for (int r = 0; r < 8; ++r) x0[r] = __shfl_xor(p0[r], 16, 32);
        #pragma unroll
        for (int r = 0; r < 8; ++r) x1[r] = __shfl_xor(p1[r], 16, 32);
        v16h pB;
        #pragma unroll
        for (int e = 0; e < 8; ++e)
            pB[e]     = (_Float16)(hf ? x1[e] : p0[e]);
        #pragma unroll
        for (int e = 0; e < 8; ++e)
            pB[8 + e] = (_Float16)(hf ? p1[e] : x0[e]);

        // out^T[c, q] += V^T[c, k] * P[k, q] ; V rows from LDS (padded
        // stride -> 16 distinct banks across the half-wave).
        #pragma unroll
        for (int tt = 0; tt < 16; ++tt) {
            const v16h av = load_a_frag(vs + (16 * tt + lm) * VSTR, o0, o1);
            acc[tt] = __builtin_amdgcn_wmma_f32_16x16x32_f16(false, av, false, pB,
                                                             (short)0, acc[tt],
                                                             false, false);
        }
    }

    // Epilogue: out = gamma * softmax_out + x. Lane holds query column
    // q0+lm; VGPR r is channel 16t + r (+8 for the upper half-wave).
    const float inv = 1.0f / l_run;
    const float g = gamma[0];
    const float* xbp = x   + (size_t)b * Cc * Nn;
    float*       ob  = out + (size_t)b * Cc * Nn;
    const int qcol = q0 + lm;
    #pragma unroll
    for (int tt = 0; tt < 16; ++tt) {
        #pragma unroll
        for (int r = 0; r < 8; ++r) {
            const int c = 16 * tt + r + hf * 8;
            const size_t idx = (size_t)c * Nn + qcol;
            ob[idx] = g * (acc[tt][r] * inv) + xbp[idx];
        }
    }
}

extern "C" void kernel_launch(void* const* d_in, const int* in_sizes, int n_in,
                              void* d_out, int out_size, void* d_ws, size_t ws_size,
                              hipStream_t stream) {
    (void)in_sizes; (void)n_in; (void)out_size; (void)ws_size;
    const float* x     = (const float*)d_in[0];
    const float* wq    = (const float*)d_in[1];
    const float* bq    = (const float*)d_in[2];
    const float* wk    = (const float*)d_in[3];
    const float* bk    = (const float*)d_in[4];
    const float* wv    = (const float*)d_in[5];
    const float* bv    = (const float*)d_in[6];
    const float* gamma = (const float*)d_in[7];
    float* out = (float*)d_out;

    // Workspace: Q (2MB) | K (2MB) | V (16MB), all f16.
    _Float16* qws = (_Float16*)d_ws;
    _Float16* kws = qws + (size_t)Bb * Nn * Dd;
    _Float16* vws = kws + (size_t)Bb * Nn * Dd;

    proj_kernel<<<dim3(Bb * Nn / 16), dim3(256), 0, stream>>>(x, wq, bq, wk, bk,
                                                              wv, bv, qws, kws, vws);
    attn_kernel<<<dim3(Bb * Nn / 128), dim3(256), 0, stream>>>(qws, kws, vws, x,
                                                               gamma, out);
}
